// DilatedAttention_83829171683592
// MI455X (gfx1250) — matrix operand order repaired and checked by reading, not compiled, not run on hardware
//
#include <hip/hip_runtime.h>
#include <hip/hip_bf16.h>

typedef __attribute__((ext_vector_type(16))) _Float16 v16h;
typedef __attribute__((ext_vector_type(8)))  float    v8f;
typedef __attribute__((ext_vector_type(4)))  unsigned int v4u;

// Problem constants
#define DMODEL 2048
#define NSEQ   8192
#define SEGW   512
#define HD     128          // head dim
#define SDIL   256          // dilated tokens per segment per head (SEG/2)
#define GSTR   136          // g row stride in halves (128 + 8 pad -> bank spread)
#define TSTR   264          // gT / P row stride in halves (256 + 8 pad)
#define SCALE  0.08838834764831845f   // 1/sqrt(128)

union ABfrag { v16h v; v4u q[2]; };
union Cfrag  { v8f  v; float f[8]; };

extern __shared__ _Float16 smem[];

__global__ __launch_bounds__(256)
void dilated_attn_kernel(const float* __restrict__ x, float* __restrict__ out)
{
    const int tid  = threadIdx.x;
    const int lane = tid & 31;
    const int wave = tid >> 5;
    const int hs   = lane >> 4;    // half-wave (0/1)
    const int ln   = lane & 15;

    const int bid = blockIdx.x;        // 1024 = B(4) * nseg(16) * H(16)
    const int h   = bid & 15;
    const int seg = (bid >> 4) & 15;
    const int b   = bid >> 8;
    const int off = h & 1;             // per-head dilation offset (h % r)

    _Float16* g  = smem;                          // [SDIL][GSTR]
    _Float16* gT = smem + SDIL * GSTR;            // [HD][TSTR]
    _Float16* Pw = gT + HD * TSTR + wave * 16 * TSTR;   // per-wave P strip [16][TSTR]

    const size_t base = ((size_t)b * NSEQ + (size_t)seg * SEGW) * DMODEL + (size_t)h * HD;
    const float* xp = x   + base;
    float*       op = out + base;

    // ---- cooperative: gather dilated rows -> f16 g + gT; zero complement rows ----
    for (int it = tid; it < SDIL * (HD / 4); it += 256) {
        const int row = it >> 5;            // 0..255 (dilated token index)
        const int c4  = (it & 31) << 2;     // 0,4,...,124
        const float4 v = *(const float4*)(xp + (size_t)(off + 2 * row) * DMODEL + c4);
        const _Float16 h0 = (_Float16)v.x, h1 = (_Float16)v.y,
                       h2 = (_Float16)v.z, h3 = (_Float16)v.w;
        _Float16* gp = g + row * GSTR + c4;
        gp[0] = h0; gp[1] = h1; gp[2] = h2; gp[3] = h3;
        gT[(c4 + 0) * TSTR + row] = h0;
        gT[(c4 + 1) * TSTR + row] = h1;
        gT[(c4 + 2) * TSTR + row] = h2;
        gT[(c4 + 3) * TSTR + row] = h3;
        // complement (non-dilated) token for this head's columns -> zero
        const float4 z = {0.f, 0.f, 0.f, 0.f};
        *(float4*)(op + (size_t)((off ^ 1) + 2 * row) * DMODEL + c4) = z;
    }
    __syncthreads();

    // ---- each wave: 2 strips of 16 query rows ----
    for (int sIdx = 0; sIdx < 2; ++sIdx) {
        const int r0 = wave * 32 + sIdx * 16;

        // Q A-fragments: 4 chunks over K=128 (16x32 f16 layout)
        ABfrag qf[4];
#pragma unroll
        for (int kc = 0; kc < 4; ++kc) {
            const _Float16* p = g + (r0 + ln) * GSTR + kc * 32 + hs * 8;
            qf[kc].q[0] = *(const v4u*)(p);
            qf[kc].q[1] = *(const v4u*)(p + 16);
        }

        // scores: 16 column tiles of 16x16, K=128 -> 4 WMMAs each
        Cfrag acc[16];
#pragma unroll
        for (int t = 0; t < 16; ++t) {
            v8f a = {};
#pragma unroll
            for (int kc = 0; kc < 4; ++kc) {
                ABfrag bf;   // B = K^T chunk: b[i] = g[key n][feat k], k in table order
                const _Float16* p = g + (t * 16 + ln) * GSTR + kc * 32 + hs * 8;
                bf.q[0] = *(const v4u*)(p);
                bf.q[1] = *(const v4u*)(p + 16);
                a = __builtin_amdgcn_wmma_f32_16x16x32_f16(
                        false, qf[kc].v, false, bf.v, (short)0, a, false, false);
            }
            acc[t].v = a * SCALE;
        }

        // ---- softmax over 256 cols, on the C-fragment layout ----
        // row M = j + hs*8 lives in vgpr j across the 16 lanes of this half-wave
#pragma unroll
        for (int j = 0; j < 8; ++j) {
            float m = acc[0].f[j];
#pragma unroll
            for (int t = 1; t < 16; ++t) m = fmaxf(m, acc[t].f[j]);
            m = fmaxf(m, __shfl_xor(m, 1, 32));
            m = fmaxf(m, __shfl_xor(m, 2, 32));
            m = fmaxf(m, __shfl_xor(m, 4, 32));
            m = fmaxf(m, __shfl_xor(m, 8, 32));
            float s = 0.f;
#pragma unroll
            for (int t = 0; t < 16; ++t) {
                const float e = __expf(acc[t].f[j] - m);
                acc[t].f[j] = e;
                s += e;
            }
            s += __shfl_xor(s, 1, 32);
            s += __shfl_xor(s, 2, 32);
            s += __shfl_xor(s, 4, 32);
            s += __shfl_xor(s, 8, 32);
            const float inv = 1.0f / s;
#pragma unroll
            for (int t = 0; t < 16; ++t)
                Pw[(j + hs * 8) * TSTR + t * 16 + ln] = (_Float16)(acc[t].f[j] * inv);
        }

        // ---- O = P (16x256) x V (256x128) ----
        ABfrag pf[8];
#pragma unroll
        for (int kc = 0; kc < 8; ++kc) {
            const _Float16* p = Pw + ln * TSTR + kc * 32 + hs * 8;
            pf[kc].q[0] = *(const v4u*)(p);
            pf[kc].q[1] = *(const v4u*)(p + 16);
        }
#pragma unroll
        for (int n0 = 0; n0 < 8; ++n0) {
            v8f oa = {};
#pragma unroll
            for (int kc = 0; kc < 8; ++kc) {
                ABfrag bf;   // B = V chunk via gT: b[i] = gT[feat n][key k], k in table order
                const _Float16* p = gT + (n0 * 16 + ln) * TSTR + kc * 32 + hs * 8;
                bf.q[0] = *(const v4u*)(p);
                bf.q[1] = *(const v4u*)(p + 16);
                oa = __builtin_amdgcn_wmma_f32_16x16x32_f16(
                        false, pf[kc].v, false, bf.v, (short)0, oa, false, false);
            }
            Cfrag o; o.v = oa;
#pragma unroll
            for (int j = 0; j < 8; ++j) {
                const int qi = r0 + j + hs * 8;   // dilated row index 0..255
                op[(size_t)(off + 2 * qi) * DMODEL + n0 * 16 + ln] = o.f[j];
            }
        }
    }
}

extern "C" void kernel_launch(void* const* d_in, const int* in_sizes, int n_in,
                              void* d_out, int out_size, void* d_ws, size_t ws_size,
                              hipStream_t stream) {
    const float* x  = (const float*)d_in[0];
    float*      out = (float*)d_out;
    // g: 256*136 halves, gT: 128*264, P: 8 waves * 16*264  -> 204800 bytes (< 320KB/WGP)
    const size_t smem_bytes =
        (size_t)(SDIL * GSTR + HD * TSTR + 8 * 16 * TSTR) * sizeof(_Float16);
    dilated_attn_kernel<<<dim3(4 * 16 * 16), dim3(256), smem_bytes, stream>>>(x, out);
}
